// HybridAttention_65481071405719
// MI455X (gfx1250) — compile-verified, matrix operand order checked by reading, not run on hardware
//
#include <hip/hip_runtime.h>
#include <hip/hip_bf16.h>
#include <stdint.h>

// ---------------------------------------------------------------------------
// HybridAttention (no softmax): out = ((x Wq^T)(x Wk^T)^T * 1/8, masked) @ (x Wv^T)
// bf16 WMMA with f32 accumulation; async-to-LDS double-buffered K/V staging.
// ---------------------------------------------------------------------------

typedef __bf16 bf16_t;
typedef bf16_t bf16x8  __attribute__((ext_vector_type(8)));
typedef bf16_t bf16x16 __attribute__((ext_vector_type(16)));
typedef float  f32x4   __attribute__((ext_vector_type(4)));
typedef float  v8f     __attribute__((ext_vector_type(8)));

#define WMMA_BF16(a, b, c) \
  __builtin_amdgcn_wmma_f32_16x16x32_bf16(false, (a), false, (b), (short)0, (c), false, false)

// A/B fragment from row-major bf16 memory (global or LDS).
// elements 0..7 = K(half8..half8+7), 8..15 = K(16+half8..), half8=(lane>>4)*8
static __device__ __forceinline__ bf16x16 ld_frag(const bf16_t* p, int half8) {
  bf16x8 lo = *(const bf16x8*)(p + half8);
  bf16x8 hi = *(const bf16x8*)(p + 16 + half8);
  bf16x16 r;
#pragma unroll
  for (int i = 0; i < 8; ++i) { r[i] = lo[i]; r[i + 8] = hi[i]; }
  return r;
}

// Async 16B copy global->LDS (CDNA5, ASYNCcnt-tracked). saddr form:
// addr = SGPR64(base) + VGPR32(byte offset); vdst supplies the LDS byte address.
static __device__ __forceinline__ void async_b128(uint32_t lds, const void* sbase,
                                                  uint32_t voff) {
  asm volatile("global_load_async_to_lds_b128 %0, %1, %2"
               :: "v"(lds), "v"(voff), "s"(sbase) : "memory");
}
// Low 32 bits of a generic LDS pointer are the LDS byte offset.
template <typename T>
static __device__ __forceinline__ uint32_t lds_off_u32(T* p) {
  return (uint32_t)(uintptr_t)p;
}

// ---------------------------------------------------------------------------
// Stage 0: one-shot fp32 -> bf16 conversion of x, wq, wk, wv (8 elems/thread)
// ---------------------------------------------------------------------------
__global__ __launch_bounds__(256) void cvt_bf16_kernel(
    const float* __restrict__ x,  const float* __restrict__ wq,
    const float* __restrict__ wk, const float* __restrict__ wv,
    bf16_t* __restrict__ xb, bf16_t* __restrict__ wbq,
    bf16_t* __restrict__ wbk, bf16_t* __restrict__ wbv) {
  const size_t NX = (size_t)4 * 2048 * 1024 / 8;   // 1M vec8 for x
  const size_t NW = (size_t)1024 * 1024 / 8;       // 128K vec8 per W
  size_t i = (size_t)blockIdx.x * blockDim.x + threadIdx.x;
  const float* src; bf16_t* dst; size_t off;
  if (i < NX)            { src = x;  dst = xb;  off = i; }
  else if (i < NX + NW)  { src = wq; dst = wbq; off = i - NX; }
  else if (i < NX + 2*NW){ src = wk; dst = wbk; off = i - NX - NW; }
  else                   { src = wv; dst = wbv; off = i - NX - 2*NW; }
  f32x4 a = *(const f32x4*)(src + off * 8);
  f32x4 b = *(const f32x4*)(src + off * 8 + 4);
  bf16x8 o;
#pragma unroll
  for (int j = 0; j < 4; ++j) { o[j] = (bf16_t)a[j]; o[4 + j] = (bf16_t)b[j]; }
  *(bf16x8*)(dst + off * 8) = o;
}

// ---------------------------------------------------------------------------
// Stage A: Q/K/V projections, pure-bf16 inner loop.
// Q,K row-major [b,h,s,dk]; V transposed [b,h,dk,s] (enables b128 stores and
// contiguous B-fragment reads in stage B). One wave = 16 tokens x 64 features.
// ---------------------------------------------------------------------------
__global__ __launch_bounds__(128) void qkv_proj_kernel(
    const bf16_t* __restrict__ xb,  const bf16_t* __restrict__ wbq,
    const bf16_t* __restrict__ wbk, const bf16_t* __restrict__ wbv,
    bf16_t* __restrict__ Qo, bf16_t* __restrict__ Ko, bf16_t* __restrict__ VTo) {
  constexpr int S = 2048, D = 1024, H = 16, DK = 64;
  const int lane  = threadIdx.x & 31;
  const int wid   = blockIdx.x * (blockDim.x >> 5) + (threadIdx.x >> 5);
  const int mat   = wid >> 13;             // 0:Q 1:K 2:V
  const int rem   = wid & 8191;
  const int ct    = rem & 15;              // 16 col tiles of 64 features
  const int rt    = rem >> 4;              // 512 row tiles of 16 tokens
  const int tbase = rt * 16, jbase = ct * 64;
  const int half8 = (lane >> 4) * 8, n15 = lane & 15;

  const bf16_t* W  = (mat == 0) ? wbq : ((mat == 1) ? wbk : wbv);
  const bf16_t* xr = xb + (size_t)(tbase + n15) * D;
  const bf16_t* wr = W  + (size_t)(jbase + n15) * D;

  v8f acc[4] = {};
#pragma unroll 2
  for (int d0 = 0; d0 < D; d0 += 32) {
    bf16x16 a = ld_frag(xr + d0, half8);
#pragma unroll
    for (int nt = 0; nt < 4; ++nt)
      acc[nt] = WMMA_BF16(a, ld_frag(wr + (size_t)(16 * nt) * D + d0, half8), acc[nt]);
  }

  const int bidx = tbase >> 11;
  const int s0   = (tbase & (S - 1)) + half8;
#pragma unroll
  for (int nt = 0; nt < 4; ++nt) {
    const int j = jbase + nt * 16 + n15;
    const int h = j >> 6, dkk = j & 63;
    if (mat < 2) {
      bf16_t* Y = (mat == 0) ? Qo : Ko;
#pragma unroll
      for (int r = 0; r < 8; ++r)
        Y[(((size_t)(bidx * H + h)) * S + (s0 + r)) * DK + dkk] = (bf16_t)acc[nt][r];
    } else {
      bf16x8 pk;                           // 8 consecutive s -> one b128 store
#pragma unroll
      for (int r = 0; r < 8; ++r) pk[r] = (bf16_t)acc[nt][r];
      *(bf16x8*)(VTo + (((size_t)(bidx * H + h)) * DK + dkk) * S + s0) = pk;
    }
  }
}

// ---------------------------------------------------------------------------
// Stage B: 4 waves/block share one (b,h); wave w owns q-tile 4*grp+w.
// K (32x64) and V^T (64x32) tiles double-buffered in LDS via async loads.
// Per 32-key iteration: 4 WMMAs (QK^T) + scale/mask + LDS transpose + 4 WMMAs.
// ---------------------------------------------------------------------------
__global__ __launch_bounds__(128) void attn_kernel(
    const bf16_t* __restrict__ Qi, const bf16_t* __restrict__ Ki,
    const bf16_t* __restrict__ VTi, const int* __restrict__ mask,
    float* __restrict__ out) {
  constexpr int S = 2048, D = 1024, DK = 64;
  __shared__ __align__(16) bf16_t ldsK[2][32 * 64];   // 2 x 4KB
  __shared__ __align__(16) bf16_t ldsV[2][64 * 32];   // 2 x 4KB
  __shared__ __align__(16) bf16_t sc[4][16 * 32];     // per-wave score staging

  const int tid   = threadIdx.x;
  const int lane  = tid & 31, wslot = tid >> 5;
  const int bh    = blockIdx.x >> 5;      // 64 (b,h) pairs
  const int qt    = (blockIdx.x & 31) * 4 + wslot;   // q-tile 0..127
  const int b     = bh >> 4, h = bh & 15;
  const int qbase = qt * 16;
  const int half8 = (lane >> 4) * 8, n15 = lane & 15;

  const bf16_t* Qb    = Qi  + (size_t)bh * S * DK;
  const bf16_t* Kb    = Ki  + (size_t)bh * S * DK;
  const bf16_t* Vb    = VTi + (size_t)bh * DK * S;
  const int*    mbase = mask + (size_t)b * S * S + (size_t)qbase * S;

  // Persistent Q fragments (dk 0..31, 32..63)
  const bf16_t* qr = Qb + (size_t)(qbase + n15) * DK;
  const bf16x16 aq0 = ld_frag(qr, half8);
  const bf16x16 aq1 = ld_frag(qr + 32, half8);

  // Cooperative async copy: 512 x 16B chunks per buffer, 4 per thread.
  const int kR0 = tid >> 3,         kC = (tid & 7) * 8;   // K rows 0..15
  const int kR1 = (tid + 128) >> 3;                       // K rows 16..31
  const int vR0 = tid >> 2,         vC = (tid & 3) * 8;   // V rows 0..31
  const int vR1 = (tid + 128) >> 2;                       // V rows 32..63
  const uint32_t lk0 = lds_off_u32(&ldsK[0][0]) + (uint32_t)(kR0 * 64 + kC) * 2;
  const uint32_t lk1 = lds_off_u32(&ldsK[0][0]) + (uint32_t)(kR1 * 64 + kC) * 2;
  const uint32_t lv0 = lds_off_u32(&ldsV[0][0]) + (uint32_t)(vR0 * 32 + vC) * 2;
  const uint32_t lv1 = lds_off_u32(&ldsV[0][0]) + (uint32_t)(vR1 * 32 + vC) * 2;

  auto issue = [&](int kb, int buf) {
    const uint32_t bo = buf ? 4096u : 0u;
    async_b128(lk0 + bo, Kb, (uint32_t)(((kb + kR0) * 64 + kC) * 2));
    async_b128(lk1 + bo, Kb, (uint32_t)(((kb + kR1) * 64 + kC) * 2));
    async_b128(lv0 + bo, Vb, (uint32_t)((vR0 * S + kb + vC) * 2));
    async_b128(lv1 + bo, Vb, (uint32_t)((vR1 * S + kb + vC) * 2));
  };

  v8f acc[4] = {};
  bf16_t* mysc = &sc[wslot][0];

  issue(0, 0);
  for (int kb = 0; kb < S; kb += 32) {
    const int buf = (kb >> 5) & 1;
    if (kb + 32 < S) {
      issue(kb + 32, buf ^ 1);                       // prefetch next tile
      asm volatile("s_wait_asynccnt 0x4" ::: "memory");  // current tile done
    } else {
      asm volatile("s_wait_asynccnt 0x0" ::: "memory");
    }
    __syncthreads();                                 // all waves' tile visible

    const bf16_t* ldk = &ldsK[buf][0];
    const bf16_t* ldv = &ldsV[buf][0];

    // ---- scores = Q K^T (2 x 16 key cols), scale + mask + stage to LDS ----
#pragma unroll
    for (int ctile = 0; ctile < 2; ++ctile) {
      const bf16_t* kr = ldk + (ctile * 16 + n15) * 64;
      v8f c = {};
      c = WMMA_BF16(aq0, ld_frag(kr, half8), c);
      c = WMMA_BF16(aq1, ld_frag(kr + 32, half8), c);
#pragma unroll
      for (int r = 0; r < 8; ++r) {
        const int m = half8 + r;
        float v = c[r] * 0.125f;
        const int mk = mbase[(size_t)m * S + kb + ctile * 16 + n15];
        v = (mk == 0) ? -1.0e9f : v;
        mysc[m * 32 + ctile * 16 + n15] = (bf16_t)v;
      }
    }
    asm volatile("s_wait_dscnt 0x0" ::: "memory");   // cross-lane LDS dep

    // ---- attn += scores @ V (4 dk tiles) ----
    const bf16x16 as = ld_frag(mysc + n15 * 32, half8);
#pragma unroll
    for (int nt = 0; nt < 4; ++nt)
      acc[nt] = WMMA_BF16(as, ld_frag(ldv + (nt * 16 + n15) * 32, half8), acc[nt]);

    __syncthreads();   // everyone done reading buf before it is re-filled
  }

  // out[b, s, h*64 + dk] (fp32)
#pragma unroll
  for (int nt = 0; nt < 4; ++nt)
#pragma unroll
    for (int r = 0; r < 8; ++r)
      out[((size_t)b * S + qbase + half8 + r) * D + h * 64 + nt * 16 + n15] = acc[nt][r];
}

// ---------------------------------------------------------------------------
extern "C" void kernel_launch(void* const* d_in, const int* in_sizes, int n_in,
                              void* d_out, int out_size, void* d_ws, size_t ws_size,
                              hipStream_t stream) {
  (void)in_sizes; (void)n_in; (void)out_size; (void)ws_size;
  const float* x    = (const float*)d_in[0];
  const int*   mask = (const int*)d_in[1];
  const float* wq   = (const float*)d_in[2];
  const float* wk   = (const float*)d_in[3];
  const float* wv   = (const float*)d_in[4];

  const size_t EXD = (size_t)4 * 2048 * 1024;   // 8M (x / Q / K / VT elems)
  const size_t EWD = (size_t)1024 * 1024;       // 1M (one W)
  bf16_t* Q   = (bf16_t*)d_ws;
  bf16_t* K   = Q   + EXD;
  bf16_t* VT  = K   + EXD;
  bf16_t* XB  = VT  + EXD;
  bf16_t* WBQ = XB  + EXD;
  bf16_t* WBK = WBQ + EWD;
  bf16_t* WBV = WBK + EWD;                      // total 35M elems = 70 MiB

  // Stage 0: fp32 -> bf16  ((8M + 3M)/8 = 1,441,792 threads)
  cvt_bf16_kernel<<<5632, 256, 0, stream>>>(x, wq, wk, wv, XB, WBQ, WBK, WBV);
  // Stage A: 24576 tile-waves / 4 per block
  qkv_proj_kernel<<<6144, 128, 0, stream>>>(XB, WBQ, WBK, WBV, Q, K, VT);
  // Stage B: 64 (b,h) x 32 blocks (4 q-tiles each)
  attn_kernel<<<2048, 128, 0, stream>>>(Q, K, VT, mask, (float*)d_out);
}
